// STGAT_44409961840694
// MI455X (gfx1250) — matrix operand order repaired and checked by reading, not compiled, not run on hardware
//
#include <hip/hip_runtime.h>
#include <hip/hip_bf16.h>
#include <math.h>

typedef __attribute__((ext_vector_type(16))) _Float16 v16h;
typedef __attribute__((ext_vector_type(8)))  float    v8f;

#define B_     8
#define N_     300
#define T_     12
#define HID_   32
#define HEADS_ 8
#define NHID_  4
#define SEQS_  (B_*N_)   // 2400
#define FOUT_  36        // T_out * F_out

// Branch-free transcendentals: v_exp_f32 + v_rcp_f32, no exec-mask churn.
__device__ __forceinline__ float sigmf_(float x){
  return __builtin_amdgcn_rcpf(1.0f + __expf(-x));
}
__device__ __forceinline__ float tanhf_(float x){
  // tanh(x) = 1 - 2/(exp(2x)+1); saturates correctly as exp -> 0 or inf.
  return 1.0f - 2.0f * __builtin_amdgcn_rcpf(__expf(2.0f*x) + 1.0f);
}

// ---------------------------------------------------------------------------
// Kernel A: g = X @ W_gat^T ; gl = g·a_l ; gr = g·a_r   (one thread per b,n,t)
// ---------------------------------------------------------------------------
__global__ void gat_embed_kernel(const float* __restrict__ X,
                                 const float* __restrict__ Wg,
                                 const float* __restrict__ a_attn,
                                 float* __restrict__ g,
                                 float* __restrict__ gl,
                                 float* __restrict__ gr)
{
  int idx = blockIdx.x * blockDim.x + threadIdx.x;
  if (idx >= B_*N_*T_) return;
  const float x0 = X[idx*3+0], x1 = X[idx*3+1], x2 = X[idx*3+2];
  float* gp = g + (size_t)idx * HID_;
  #pragma unroll
  for (int h = 0; h < HEADS_; ++h){
    float ga = 0.f, gb = 0.f;
    #pragma unroll
    for (int d = 0; d < NHID_; ++d){
      int c = h*NHID_ + d;
      float gv = x0*Wg[c*3+0] + x1*Wg[c*3+1] + x2*Wg[c*3+2];
      gp[c] = gv;
      ga += gv * a_attn[d];
      gb += gv * a_attn[NHID_ + d];
    }
    gl[(size_t)idx*HEADS_ + h] = ga;
    gr[(size_t)idx*HEADS_ + h] = gb;
  }
}

// ---------------------------------------------------------------------------
// Kernel B: masked attention softmax + aggregation (online softmax).
// Block = (b,i); 96 threads = (t,h) pairs; A_hat row staged in LDS.
// ---------------------------------------------------------------------------
__global__ __launch_bounds__(96)
void gat_attn_kernel(const float* __restrict__ A_hat,
                     const float* __restrict__ g,
                     const float* __restrict__ gl,
                     const float* __restrict__ gr,
                     float* __restrict__ hgat)
{
  __shared__ float Arow[N_];
  int bi = blockIdx.x;            // b*N + i
  int b  = bi / N_;
  for (int j = threadIdx.x; j < N_; j += 96)
    Arow[j] = A_hat[(size_t)(bi - b*N_)*N_ + j];
  __syncthreads();

  int t = threadIdx.x >> 3;
  int h = threadIdx.x & 7;
  float glv = gl[((size_t)bi*T_ + t)*HEADS_ + h];
  float m = -1e30f, s = 0.f;
  float a0 = 0.f, a1 = 0.f, a2 = 0.f, a3 = 0.f;
  for (int j = 0; j < N_; ++j){
    if (Arow[j] != 0.f){
      size_t jb = (size_t)(b*N_ + j)*T_ + t;
      float e = glv + gr[jb*HEADS_ + h];
      e = (e > 0.f) ? e : 0.2f*e;                 // leaky_relu
      float w;
      if (e > m){
        float sc = __expf(m - e);
        s *= sc; a0 *= sc; a1 *= sc; a2 *= sc; a3 *= sc;
        m = e; w = 1.f;
      } else {
        w = __expf(e - m);
      }
      const float* gp = g + jb*HID_ + h*NHID_;
      s  += w;
      a0 += w*gp[0]; a1 += w*gp[1]; a2 += w*gp[2]; a3 += w*gp[3];
    }
  }
  float inv = 1.f/s;                              // diag of A_hat guarantees s>0
  float* op = hgat + ((size_t)bi*T_ + t)*HID_ + h*NHID_;
  op[0]=a0*inv; op[1]=a1*inv; op[2]=a2*inv; op[3]=a3*inv;
}

// ---------------------------------------------------------------------------
// Kernel C: pack the four (128x32) LSTM weight matrices into f16 WMMA B-layout.
// B tile n covers output cols [16n,16n+16): lane 0-15 -> col, K=0..15 in the
// 16 halves; lane 16-31 -> col, K=16..31.   dst[(m*8+tile)*32+lane][0..15]
// ---------------------------------------------------------------------------
__global__ __launch_bounds__(32)
void pack_w_kernel(const float* __restrict__ Wih0, const float* __restrict__ Whh0,
                   const float* __restrict__ Wih1, const float* __restrict__ Whh1,
                   _Float16* __restrict__ wpack)
{
  int m    = blockIdx.x >> 3;     // 0..3 : Wih0, Whh0, Wih1, Whh1
  int tile = blockIdx.x & 7;      // 0..7 : column tile
  const float* W = (m==0) ? Wih0 : (m==1) ? Whh0 : (m==2) ? Wih1 : Whh1;
  int lane = threadIdx.x;
  int col  = 16*tile + (lane & 15);
  int kb   = (lane >> 4) * 16;
  _Float16* dst = wpack + (((size_t)m*8 + tile)*32 + lane)*16;
  #pragma unroll
  for (int e = 0; e < 16; ++e)
    dst[e] = (_Float16)W[(size_t)col*HID_ + kb + e];   // B[k][col] = W[col][k]
}

// ---------------------------------------------------------------------------
// Kernel D: fused 2-layer LSTM (T=12) + FC, one wave per 16 sequences.
// Per step per layer: 8 output tiles, each tile = bias -> WMMA(xt,Wih) ->
// WMMA(h,Whh) chained on the f32 accumulator; gate nonlinearity in VALU
// (branch-free exp/rcp); h re-laid-out through LDS (D-layout store ->
// A-layout f16 gather).
// ---------------------------------------------------------------------------
__global__ __launch_bounds__(32)
void lstm_fc_kernel(const float* __restrict__ xin,        // [2400][12][32]
                    const _Float16* __restrict__ wpack,   // [4][8][32][16]
                    const float* __restrict__ bih0, const float* __restrict__ bhh0,
                    const float* __restrict__ bih1, const float* __restrict__ bhh1,
                    const float* __restrict__ fcW, const float* __restrict__ fcb,
                    float* __restrict__ out)
{
  __shared__ float h1buf[T_][16][32];   // layer-0 hidden trajectory (24 KB)
  __shared__ float hstage[16][32];      // current h for A-operand rebuild
  const int lane = threadIdx.x;
  const int r    = lane & 15;
  const int hi   = lane >> 4;
  const int kb   = hi * 8;              // unused; A operands use K base 16*hi
  const int seq0 = blockIdx.x * 16;

  for (int layer = 0; layer < 2; ++layer){
    const _Float16* Wih = wpack + (size_t)layer*2*8*32*16;
    const _Float16* Whh = Wih + 8*32*16;
    const float* bih = layer ? bih1 : bih0;
    const float* bhh = layer ? bhh1 : bhh0;

    float bcol[8];
    v16h wA[8], wB[8];
    #pragma unroll
    for (int n = 0; n < 8; ++n){
      bcol[n] = bih[16*n + r] + bhh[16*n + r];
      wA[n] = *(const v16h*)(Wih + ((size_t)n*32 + lane)*16);
      wB[n] = *(const v16h*)(Whh + ((size_t)n*32 + lane)*16);
    }

    for (int q = lane; q < 16*32; q += 32) (&hstage[0][0])[q] = 0.f;
    __syncthreads();
    v8f cst[2] = {};                    // cell state, D-layout

    for (int t = 0; t < T_; ++t){
      // Build A operands (16x32 f16): lane<16 -> row r, K {0..15};
      // lane>=16 -> row r, K {16..31}.
      v16h xa, ha;
      const float* xrow = layer ? &h1buf[t][r][0]
                                : (xin + ((size_t)(seq0 + r)*T_ + t)*HID_);
      const float* hrow = &hstage[r][0];
      #pragma unroll
      for (int e = 0; e < 16; ++e){
        int k = 16*hi + e;
        xa[e] = (_Float16)xrow[k];
        ha[e] = (_Float16)hrow[k];
      }

      v8f z[8];
      #pragma unroll
      for (int n = 0; n < 8; ++n){
        float bv = bcol[n];
        v8f c = {bv,bv,bv,bv,bv,bv,bv,bv};
        c = __builtin_amdgcn_wmma_f32_16x16x32_f16(false, xa, false, wA[n],
                                                   (short)0, c, false, false);
        c = __builtin_amdgcn_wmma_f32_16x16x32_f16(false, ha, false, wB[n],
                                                   (short)0, c, false, false);
        z[n] = c;
      }

      __syncthreads();
      // D-layout element (rr,lane): row = rr+8*hi, col(within tile) = r.
      // Gates: tiles 0-1 = i, 2-3 = f, 4-5 = g, 6-7 = o.
      #pragma unroll
      for (int p = 0; p < 2; ++p){
        #pragma unroll
        for (int rr = 0; rr < 8; ++rr){
          float iv = sigmf_(z[0+p][rr]);
          float fv = sigmf_(z[2+p][rr]);
          float gv = tanhf_(z[4+p][rr]);
          float ov = sigmf_(z[6+p][rr]);
          float cc = fv*cst[p][rr] + iv*gv;
          cst[p][rr] = cc;
          float hh = ov*tanhf_(cc);
          int row = rr + 8*hi, col = 16*p + r;
          hstage[row][col] = hh;
          if (layer == 0) h1buf[t][row][col] = hh;
        }
      }
      __syncthreads();
    }
    __syncthreads();
  }

  // FC: out = h_last @ fcW^T + fcb   (h_last is in hstage)
  for (int idx = lane; idx < 16*FOUT_; idx += 32){
    int s = idx / FOUT_;
    int j = idx - s*FOUT_;
    const float* wr = fcW + (size_t)j*HID_;
    float acc = fcb[j];
    #pragma unroll
    for (int k = 0; k < HID_; ++k) acc += hstage[s][k]*wr[k];
    out[(size_t)(seq0 + s)*FOUT_ + j] = acc;
  }
}

// ---------------------------------------------------------------------------
extern "C" void kernel_launch(void* const* d_in, const int* in_sizes, int n_in,
                              void* d_out, int out_size, void* d_ws, size_t ws_size,
                              hipStream_t stream)
{
  (void)in_sizes; (void)n_in; (void)out_size; (void)ws_size;
  const float* A_hat = (const float*)d_in[0];
  const float* X     = (const float*)d_in[1];
  const float* W_gat = (const float*)d_in[2];
  const float* a_att = (const float*)d_in[3];
  const float* Wih0  = (const float*)d_in[4];
  const float* Whh0  = (const float*)d_in[5];
  const float* bih0  = (const float*)d_in[6];
  const float* bhh0  = (const float*)d_in[7];
  const float* Wih1  = (const float*)d_in[8];
  const float* Whh1  = (const float*)d_in[9];
  const float* bih1  = (const float*)d_in[10];
  const float* bhh1  = (const float*)d_in[11];
  const float* fcW   = (const float*)d_in[12];
  const float* fcb   = (const float*)d_in[13];

  // Workspace layout (all offsets 32B-aligned): ~9.25 MB total, L2-resident.
  float* g    = (float*)d_ws;                           // 921600 f32
  float* gl   = g  + (size_t)B_*N_*T_*HID_;             // 230400 f32
  float* gr   = gl + (size_t)B_*N_*T_*HEADS_;           // 230400 f32
  float* hgat = gr + (size_t)B_*N_*T_*HEADS_;           // 921600 f32
  _Float16* wpack = (_Float16*)(hgat + (size_t)B_*N_*T_*HID_);  // 16384 f16

  int bnt = B_*N_*T_;
  gat_embed_kernel<<<(bnt + 255)/256, 256, 0, stream>>>(X, W_gat, a_att, g, gl, gr);
  gat_attn_kernel<<<B_*N_, 96, 0, stream>>>(A_hat, g, gl, gr, hgat);
  pack_w_kernel<<<32, 32, 0, stream>>>(Wih0, Whh0, Wih1, Whh1, wpack);
  lstm_fc_kernel<<<SEQS_/16, 32, 0, stream>>>(hgat, wpack, bih0, bhh0,
                                              bih1, bhh1, fcW, fcb, (float*)d_out);
}
